// Convalg_45715631898899
// MI455X (gfx1250) — compile-verified, matrix operand order checked by reading, not run on hardware
//
#include <hip/hip_runtime.h>
#include <cstdint>
#include <cstddef>

// ---------------------------------------------------------------------------
// IR-Net binary conv block for MI455X (gfx1250, wave32).
// Engine: V_WMMA_I32_16X16X64_IU8 on sign-encoded int8 operands.
// Data path: GLOBAL_LOAD_ASYNC_TO_LDS_B128 staging -> DS_LOAD_B128 fragments.
// ---------------------------------------------------------------------------

typedef int v8i __attribute__((ext_vector_type(8)));

#define NB   32          // batch
#define CI_  256         // input channels
#define CO_  256         // output channels
#define HW_  56          // spatial
#define HP_  58          // padded spatial (1-halo each side)
#define KTOT 2304        // 256*3*3
#define CNT_PER_CH (NB * HW_ * HW_)     // 100352 BN population per channel

#define XT_ROWS 4                        // padded rows staged per block
#define XT_MAIN (XT_ROWS * HP_ * CI_)    // 59392 bytes staged
#define XT_BYTES (XT_MAIN + 2048)        // + pad for ragged w-tile reads

// ---------------------------------------------------------------------------
// Kernel 1: binarize activations NCHW fp32 -> zero-padded NHWC int8 signs.
// ---------------------------------------------------------------------------
__global__ void __launch_bounds__(256) binarize_x_kernel(
    const float* __restrict__ x, int8_t* __restrict__ xs) {
  __shared__ int8_t tile[CI_ * HW_];            // 14336 B
  const int nh = blockIdx.x;                    // 0 .. NB*HW-1
  const int n = nh / HW_, h = nh % HW_;
  const int t = threadIdx.x;
  const float* xrow = x + (size_t)n * CI_ * HW_ * HW_ + (size_t)h * HW_;
  for (int j = t; j < CI_ * HW_; j += 256) {    // read: w-contiguous per c
    const int c = j / HW_, w = j % HW_;
    const float v = xrow[(size_t)c * HW_ * HW_ + w];
    tile[c * HW_ + w] = (int8_t)((v > 0.f) - (v < 0.f));   // sign(0)=0, like jnp.sign
  }
  __syncthreads();
  int8_t* orow = xs + (((size_t)n * HP_ + (h + 1)) * HP_ + 1) * CI_;
  for (int j = t; j < HW_ * CI_; j += 256) {    // write: c-contiguous (NHWC)
    const int w = j / CI_, c = j % CI_;
    orow[(size_t)w * CI_ + c] = tile[c * HW_ + w];
  }
}

// ---------------------------------------------------------------------------
// Kernel 2: weight standardization + sign + power-of-2 scale, packed into
// per-lane 32-byte A-fragments for the IU8 WMMA (lane = khalf*16 + (co&15),
// lane bytes j=0..31 cover ci = cc*64 + khalf*32 + j).
// ---------------------------------------------------------------------------
__global__ void __launch_bounds__(256) prep_w_kernel(
    const float* __restrict__ w, int8_t* __restrict__ wpk,
    float* __restrict__ sw) {
  __shared__ float red[256];
  const int co = blockIdx.x;
  const int t = threadIdx.x;
  const float* wc = w + (size_t)co * KTOT;
  float lv[9];
  float s = 0.f;
#pragma unroll
  for (int k = 0; k < 9; ++k) { lv[k] = wc[t + k * 256]; s += lv[k]; }
  red[t] = s; __syncthreads();
  for (int st = 128; st > 0; st >>= 1) { if (t < st) red[t] += red[t + st]; __syncthreads(); }
  const float mean = red[0] * (1.0f / KTOT);
  __syncthreads();
  float ssq = 0.f, sab = 0.f;
#pragma unroll
  for (int k = 0; k < 9; ++k) { const float d = lv[k] - mean; ssq += d * d; sab += fabsf(d); }
  red[t] = ssq; __syncthreads();
  for (int st = 128; st > 0; st >>= 1) { if (t < st) red[t] += red[t + st]; __syncthreads(); }
  const float stdv = sqrtf(red[0] * (1.0f / (KTOT - 1)));   // torch unbiased std
  __syncthreads();
  red[t] = sab; __syncthreads();
  for (int st = 128; st > 0; st >>= 1) { if (t < st) red[t] += red[t + st]; __syncthreads(); }
  const float mabs = (red[0] * (1.0f / KTOT)) / stdv;       // mean|bw|
  if (t == 0) sw[co] = exp2f(rintf(log2f(mabs)));           // detached 2^round(log2)

  const int cotile = co >> 4, cloc = co & 15;
#pragma unroll
  for (int k = 0; k < 9; ++k) {
    const int e = t + k * 256;                 // flat index within channel (ci*9+kh*3+kw)
    const int ci = e / 9, r = e % 9, kh = r / 3, kw = r % 3;
    const int cc = ci >> 6, rem = ci & 63, khalf = rem >> 5, j = rem & 31;
    const float d = lv[k] - mean;
    const int8_t sgn = (int8_t)((d > 0.f) - (d < 0.f));
    const size_t off = ((((size_t)cotile * 9 + kh * 3 + kw) * 4 + cc) * 1024)
                     + (size_t)(khalf * 16 + cloc) * 32 + j;
    wpk[off] = sgn;
  }
}

// ---------------------------------------------------------------------------
// Kernel 3: implicit-GEMM binary conv.
// Block = 8 waves, covers ALL 256 co for 2 output rows x 64 w-columns.
// Stage: async-DMA 4 padded NHWC rows (59392 B) into LDS (ASYNCcnt).
// Wave (wv): row = wv&1, w-tile = wv>>1; loops 16 co-tiles in chunks of 4
// accumulators so each B-fragment (ds_load) feeds 4 WMMAs.
// ---------------------------------------------------------------------------
__global__ void __launch_bounds__(256) conv_wmma_kernel(
    const int8_t* __restrict__ xs, const int8_t* __restrict__ wpk,
    const float* __restrict__ sw, float* __restrict__ out) {
  __shared__ __align__(64) int8_t xtile[XT_BYTES];
  const int t = threadIdx.x;
  const int h0 = blockIdx.x * 2;         // 0,2,..,54
  const int n  = blockIdx.y;             // 0..31
  const int wv = t >> 5, lane = t & 31;
  const int nn = lane & 15, khalf = lane >> 4;

  __builtin_prefetch(wpk, 0, 3);         // -> global_prefetch_b8 (weight stream)

  // ---- async staging: padded rows h0..h0+3, NHWC signs -> LDS ----
  const uint64_t gbase = (uint64_t)(uintptr_t)(xs + ((size_t)n * HP_ + h0) * HP_ * CI_);
  const uint32_t lbase = (uint32_t)(uintptr_t)xtile;   // LDS byte offset (addr[31:0])
  const int NCHUNK = XT_MAIN / 16;                     // 3712 x 16B
  for (int i = t; i < NCHUNK; i += 256) {
    const uint32_t lo = lbase + (uint32_t)i * 16u;
    const uint32_t go = (uint32_t)i * 16u;
    asm volatile("global_load_async_to_lds_b128 %0, %1, %2"
                 :: "v"(lo), "v"(go), "s"(gbase) : "memory");
  }
  asm volatile("s_wait_asynccnt 0x0" ::: "memory");
  __syncthreads();

  const int row = wv & 1;                // output row within strip
  const int w0  = (wv >> 1) * 16;        // 0,16,32,48 (48-tile ragged, masked at store)
  const int h   = h0 + row;
  const int wcol = w0 + nn;
  const bool valid = (wcol < HW_);

  for (int cchunk = 0; cchunk < 4; ++cchunk) {         // 4 x 4 co-tiles
    v8i acc0 = {}, acc1 = {}, acc2 = {}, acc3 = {};
    const int8_t* wb = wpk + (size_t)(cchunk * 4) * (9 * 4 * 1024)
                           + (size_t)lane * 32;
#pragma unroll
    for (int kh = 0; kh < 3; ++kh) {
#pragma unroll
      for (int kw = 0; kw < 3; ++kw) {
#pragma unroll
        for (int cc = 0; cc < 4; ++cc) {
          const int boff = ((row + kh) * HP_ + (w0 + nn + kw)) * CI_
                         + cc * 64 + khalf * 32;
          const v8i b = *(const v8i*)(xtile + boff);    // ds_load_b128 x2
          const size_t aoff = (size_t)((kh * 3 + kw) * 4 + cc) * 1024;
          v8i a;
          a = *(const v8i*)(wb + aoff);
          acc0 = __builtin_amdgcn_wmma_i32_16x16x64_iu8(true, a, true, b, acc0, false, false);
          a = *(const v8i*)(wb + 1 * 36864 + aoff);
          acc1 = __builtin_amdgcn_wmma_i32_16x16x64_iu8(true, a, true, b, acc1, false, false);
          a = *(const v8i*)(wb + 2 * 36864 + aoff);
          acc2 = __builtin_amdgcn_wmma_i32_16x16x64_iu8(true, a, true, b, acc2, false, false);
          a = *(const v8i*)(wb + 3 * 36864 + aoff);
          acc3 = __builtin_amdgcn_wmma_i32_16x16x64_iu8(true, a, true, b, acc3, false, false);
        }
      }
    }
    if (valid) {
      const size_t obase = (size_t)n * CO_ * HW_ * HW_ + (size_t)h * HW_ + wcol;
#pragma unroll
      for (int r = 0; r < 8; ++r) {
        const int m = r + 8 * khalf;     // i32 C/D layout: M = r + 8*(lane>=16)
        int co = (cchunk * 4) * 16 + m;
        out[obase + (size_t)co * (HW_ * HW_)] = (float)acc0[r] * sw[co]; co += 16;
        out[obase + (size_t)co * (HW_ * HW_)] = (float)acc1[r] * sw[co]; co += 16;
        out[obase + (size_t)co * (HW_ * HW_)] = (float)acc2[r] * sw[co]; co += 16;
        out[obase + (size_t)co * (HW_ * HW_)] = (float)acc3[r] * sw[co];
      }
    }
  }
}

// ---------------------------------------------------------------------------
// Kernel 4: deterministic per-channel BN batch statistics + affine fold.
// ---------------------------------------------------------------------------
__global__ void __launch_bounds__(256) chan_stats_kernel(
    const float* __restrict__ y, const float* __restrict__ gamma,
    const float* __restrict__ beta, float* __restrict__ scale,
    float* __restrict__ bias) {
  __shared__ float rs[256], rq[256];
  const int c = blockIdx.x;
  const int t = threadIdx.x;
  float s = 0.f, q = 0.f;
  for (int n = 0; n < NB; ++n) {
    const float* p = y + ((size_t)n * CO_ + c) * (HW_ * HW_);
    for (int i = t; i < HW_ * HW_; i += 256) {
      const float v = p[i];
      s += v; q += v * v;
    }
  }
  rs[t] = s; rq[t] = q; __syncthreads();
  for (int st = 128; st > 0; st >>= 1) {
    if (t < st) { rs[t] += rs[t + st]; rq[t] += rq[t + st]; }
    __syncthreads();
  }
  if (t == 0) {
    const float inv = 1.0f / (float)CNT_PER_CH;
    const float mean = rs[0] * inv;
    const float var = rq[0] * inv - mean * mean;     // biased, like jnp.var
    const float sc = gamma[c] * rsqrtf(var + 1e-5f);
    scale[c] = sc;
    bias[c] = beta[c] - mean * sc;
  }
}

// ---------------------------------------------------------------------------
// Kernel 5: BN affine + hardtanh, in place on d_out.
// ---------------------------------------------------------------------------
__global__ void __launch_bounds__(256) bn_apply_kernel(
    float* __restrict__ out, const float* __restrict__ scale,
    const float* __restrict__ bias, size_t total) {
  const size_t i = (size_t)blockIdx.x * 256 + threadIdx.x;
  if (i >= total) return;
  const int c = (int)((i / (HW_ * HW_)) % CO_);
  const float v = out[i] * scale[c] + bias[c];
  out[i] = fminf(1.f, fmaxf(-1.f, v));
}

// ---------------------------------------------------------------------------
extern "C" void kernel_launch(void* const* d_in, const int* in_sizes, int n_in,
                              void* d_out, int out_size, void* d_ws, size_t ws_size,
                              hipStream_t stream) {
  (void)in_sizes; (void)n_in; (void)out_size; (void)ws_size;
  const float* x     = (const float*)d_in[0];
  const float* wt    = (const float*)d_in[1];
  const float* gamma = (const float*)d_in[2];
  const float* beta  = (const float*)d_in[3];
  float* out = (float*)d_out;
  uint8_t* ws = (uint8_t*)d_ws;

  // workspace layout (all 256B-aligned)
  const size_t XS_BYTES = (size_t)NB * HP_ * HP_ * CI_;   // 27,557,888
  const size_t XS_SLACK = 4096;
  const size_t WPK_OFF  = XS_BYTES + XS_SLACK;
  const size_t WPK_SZ   = (size_t)16 * 9 * 4 * 1024;      // 589,824
  const size_t SW_OFF   = WPK_OFF + WPK_SZ;
  const size_t SCL_OFF  = SW_OFF + 1024;
  const size_t BIA_OFF  = SCL_OFF + 1024;

  int8_t* xs    = (int8_t*)ws;
  int8_t* wpk   = (int8_t*)(ws + WPK_OFF);
  float*  sw    = (float*)(ws + SW_OFF);
  float*  scale = (float*)(ws + SCL_OFF);
  float*  bias  = (float*)(ws + BIA_OFF);

  // zero padded-sign buffer (halo must be 0 so padding contributes nothing)
  hipMemsetAsync(xs, 0, XS_BYTES + XS_SLACK, stream);

  binarize_x_kernel<<<NB * HW_, 256, 0, stream>>>(x, xs);
  prep_w_kernel<<<CO_, 256, 0, stream>>>(wt, wpk, sw);

  dim3 cg(HW_ / 2, NB);   // 28 row-strips x 32 images; block covers all 256 co
  conv_wmma_kernel<<<cg, 256, 0, stream>>>(xs, wpk, sw, out);

  chan_stats_kernel<<<CO_, 256, 0, stream>>>(out, gamma, beta, scale, bias);

  const size_t total = (size_t)NB * CO_ * HW_ * HW_;
  bn_apply_kernel<<<(unsigned)((total + 255) / 256), 256, 0, stream>>>(
      out, scale, bias, total);
}
